// RefineVitLayer_24644522344932
// MI455X (gfx1250) — matrix operand order repaired and checked by reading, not compiled, not run on hardware
//
#include <hip/hip_runtime.h>
#include <hip/hip_fp16.h>
#include <stdint.h>

typedef __attribute__((ext_vector_type(16))) _Float16 v16h;
typedef __attribute__((ext_vector_type(8)))  float    v8f;

#define WMMA_F16(A, B, Cacc) \
  __builtin_amdgcn_wmma_f32_16x16x32_f16(false, (A), false, (B), (short)0, (Cacc), false, false)

// ---- CDNA5 async global->LDS staging (ASYNCcnt), guarded so compile never breaks
#if defined(__gfx1250__) && __has_builtin(__builtin_amdgcn_global_load_async_to_lds_b128) && \
    __has_builtin(__builtin_amdgcn_s_wait_asynccnt)
#define HAVE_ASYNC_LDS 1
typedef int v4i_vs __attribute__((vector_size(16)));   // matches builtin param type
#define STAGE_WAIT(n) __builtin_amdgcn_s_wait_asynccnt(n)
#else
#define HAVE_ASYNC_LDS 0
#define STAGE_WAIT(n) ((void)0)
#endif

__device__ __forceinline__ void async_cp16(const void* g, void* l) {
#if HAVE_ASYNC_LDS
  __builtin_amdgcn_global_load_async_to_lds_b128((v4i_vs*)g, (v4i_vs*)l, 0, 0);
#else
  *(uint4*)l = *(const uint4*)g;
#endif
}
// copy `rows` rows of rowB bytes (16B multiple) global->LDS; strides in bytes
__device__ __forceinline__ void stage_rows(const char* g, int gldB, char* l, int lldB,
                                           int rows, int rowB, int tid) {
  int cpr = rowB >> 4;
  int total = rows * cpr;
  for (int i = tid; i < total; i += 256) {
    int r = i / cpr, ch = (i % cpr) << 4;
    async_cp16(g + (size_t)r * gldB + ch, l + (size_t)r * lldB + ch);
  }
}

// ---------------------------------------------------------------------------
// WMMA f16 fragment loaders; A and B both use contiguous k = 16*(lane/16) + s.
// ---------------------------------------------------------------------------
__device__ __forceinline__ v16h frag_a(const _Float16* A, int ld, int m0, int k0, int lane) {
  v16h f;
  int lh = (lane & 16);
  const _Float16* r = A + (size_t)(m0 + (lane & 15)) * ld;
#pragma unroll
  for (int j = 0; j < 8; ++j) {
    f[j]     = r[k0 + lh + j];
    f[j + 8] = r[k0 + 8 + lh + j];
  }
  return f;
}
__device__ __forceinline__ v16h frag_b(const _Float16* B, int ld, int k0, int n0, int lane) {
  v16h f;
  int kh = (lane & 16);
  const _Float16* c = B + n0 + (lane & 15);
#pragma unroll
  for (int j = 0; j < 16; ++j) f[j] = c[(size_t)(k0 + kh + j) * ld];
  return f;
}
__device__ __forceinline__ v16h frag_bt(const _Float16* T, int ld, int k0, int n0, int lane) {
  v16h f;
  int kh = (lane & 16);
  const _Float16* r = T + (size_t)(n0 + (lane & 15)) * ld + k0 + kh;
#pragma unroll
  for (int j = 0; j < 16; ++j) f[j] = r[j];
  return f;
}
__device__ __forceinline__ float gelu_f(float v) {
  return 0.5f * v * (1.0f + erff(v * 0.70710678118654752f));
}

// ---------------------------------------------------------------------------
// Weight pre-conversion (once per call): f32 -> f16 in workspace.
// ---------------------------------------------------------------------------
__global__ void cvt_f16_kernel(const float* __restrict__ src, _Float16* __restrict__ dst, int n) {
  int i = blockIdx.x * 256 + threadIdx.x;
  if (i < n) dst[i] = (_Float16)src[i];
}
// pointwise conv weights transposed: dst[cin*64+cout] = pw[cout*64+cin]
__global__ void cvt_pw_kernel(const float* __restrict__ pw, _Float16* __restrict__ dst) {
  int i = blockIdx.x * 256 + threadIdx.x;   // 4096
  dst[i] = (_Float16)pw[(i & 63) * 64 + (i >> 6)];
}

// ---------------------------------------------------------------------------
// Kernel 1: SeparableConvBNReLU(6) [+ optional residual].
// 8x8 tile x 64ch per WG; depthwise fp32 -> f16 A in LDS -> pointwise as
// 64x64x64 WMMA -> BN+clip (+res) staged in LDS -> coalesced NCHW store.
// ---------------------------------------------------------------------------
__global__ __launch_bounds__(256) void sepconv_kernel(
    const float* __restrict__ xin, const float* __restrict__ dww,
    const _Float16* __restrict__ pw16, const float* __restrict__ bng,
    const float* __restrict__ bnb, float* __restrict__ out, int addResidual)
{
  extern __shared__ char smem[];
  float*    sin  = (float*)smem;                        // [196][64]
  float*    sdw  = (float*)(smem + 50176);              // [64][49]
  _Float16* aH   = (_Float16*)(smem + 62720);           // [64 pix][64 cin]
  _Float16* bH   = (_Float16*)(smem + 70912);           // [64 cin][64 cout]
  float*    outf = (float*)(smem + 79104);              // [64 pix][64 cout]

  const int tid  = threadIdx.x;
  const int wave = tid >> 5, lane = tid & 31;
  const int b  = blockIdx.x >> 12;
  const int t  = blockIdx.x & 4095;
  const int ty = (t >> 6) * 8, tx = (t & 63) * 8;
  const size_t cstride = (size_t)512 * 512;
  const float resScale = (float)addResidual;

  // async-stage weights into LDS (overlaps with halo-tile load below)
  stage_rows((const char*)pw16, 8192, (char*)bH, 8192, 1, 8192, tid);
  stage_rows((const char*)dww, 12544, (char*)sdw, 12544, 1, 12544, tid);

  for (int i = tid; i < 196 * 64; i += 256) {
    int c = i / 196, p = i % 196;
    int y = ty + (p / 14) - 3, x = tx + (p % 14) - 3;
    float v = 0.f;
    if ((unsigned)y < 512u && (unsigned)x < 512u)
      v = xin[((size_t)b * 64 + c) * cstride + (size_t)y * 512 + x];
    sin[p * 64 + c] = v;
  }
  STAGE_WAIT(0);
  __syncthreads();

  // depthwise 7x7 (zero-padded), f16 A matrix [pixel][cin]
  for (int i = tid; i < 64 * 64; i += 256) {
    int c = i & 63, p = i >> 6;
    int py = p >> 3, px = p & 7;
    float acc = 0.f;
#pragma unroll
    for (int dy = 0; dy < 7; ++dy)
#pragma unroll
      for (int dx = 0; dx < 7; ++dx)
        acc += sin[((py + dy) * 14 + (px + dx)) * 64 + c] * sdw[c * 49 + dy * 7 + dx];
    aH[p * 64 + c] = (_Float16)acc;
  }
  __syncthreads();

  // pointwise 1x1: M=64 pix, N=64 cout, K=64 cin
  for (int tt = wave; tt < 16; tt += 8) {
    int mt = tt >> 2, nt = tt & 3;
    v8f acc = {};
#pragma unroll
    for (int ks = 0; ks < 2; ++ks) {
      v16h af = frag_a(aH, 64, mt * 16, ks * 32, lane);
      v16h bf = frag_b(bH, 64, ks * 32, nt * 16, lane);
      acc = WMMA_F16(af, bf, acc);
    }
    int n = nt * 16 + (lane & 15);
    float scale = bng[n] * 0.99999500003749969f;        // g / sqrt(1+1e-5)
    float bias  = bnb[n];
    int mbase = mt * 16 + ((lane & 16) >> 1);
#pragma unroll
    for (int r = 0; r < 8; ++r)
      outf[(mbase + r) * 64 + n] = fminf(fmaxf(acc[r] * scale + bias, 0.f), 6.f);
  }
  __syncthreads();

  // fused residual + coalesced store (consecutive tids -> consecutive x)
  for (int i = tid; i < 4096; i += 256) {
    int c = i >> 6, p = i & 63;
    int py = p >> 3, px = p & 7;
    float v = outf[p * 64 + c] + resScale * sin[((py + 3) * 14 + (px + 3)) * 64 + c];
    out[((size_t)b * 64 + c) * cstride + (size_t)(ty + py) * 512 + (tx + px)] = v;
  }
}

// ---------------------------------------------------------------------------
// Kernel 2: per-window uncertainty mean (zero-padded 8x8)
// ---------------------------------------------------------------------------
__global__ void score_kernel(const float* __restrict__ unc, float* __restrict__ score,
                             int nH, int nW, int padT, int padL)
{
  __shared__ float red[64];
  const int nWin = nH * nW;
  const int wid = blockIdx.x % nWin, b = blockIdx.x / nWin;
  const int tid = threadIdx.x;
  int y = (wid / nW) * 8 - padT + (tid >> 3);
  int x = (wid % nW) * 8 - padL + (tid & 7);
  float v = 0.f;
  if ((unsigned)y < 512u && (unsigned)x < 512u)
    v = unc[((size_t)b * 512 + y) * 512 + x];
  red[tid] = v;
  __syncthreads();
  for (int s = 32; s > 0; s >>= 1) {
    if (tid < s) red[tid] += red[tid + s];
    __syncthreads();
  }
  if (tid == 0) score[(size_t)b * nWin + wid] = red[0] * (1.f / 64.f);
}

// ---------------------------------------------------------------------------
// Kernel 3: exact top-k by rank counting (tie -> lower index, matches top_k)
// ---------------------------------------------------------------------------
__global__ void select_kernel(const float* __restrict__ score, int* __restrict__ sel,
                              int nWin, int nWF)
{
  extern __shared__ float sc[];
  const int b = blockIdx.x;
  for (int i = threadIdx.x; i < nWin; i += blockDim.x) sc[i] = score[(size_t)b * nWin + i];
  __syncthreads();
  for (int i = threadIdx.x; i < nWin; i += blockDim.x) {
    float s = sc[i];
    int rank = 0;
    for (int j = 0; j < nWin; ++j) {
      float t = sc[j];
      rank += (t > s) || (t == s && j < i);
    }
    if (rank < nWF) sel[(size_t)b * nWF + rank] = i;
  }
}

// ---------------------------------------------------------------------------
// Kernel 4: refine one selected 8x8 window. f16 weights pre-converted in ws;
// async double-buffered LDS staging pipelined against WMMA GEMMs.
// ---------------------------------------------------------------------------
__global__ __launch_bounds__(256) void refine_kernel(
    float* __restrict__ xg, const int* __restrict__ sel,
    int nW, int nWF, int padT, int padL,
    const float* __restrict__ lng,  const float* __restrict__ lnb,
    const _Float16* __restrict__ lin16, const float* __restrict__ linb,
    const _Float16* __restrict__ qkv16, const _Float16* __restrict__ proj16,
    const float* __restrict__ projb)
{
  extern __shared__ char smem[];
  _Float16* xfh  = (_Float16*)smem;                     // [64][128]
  _Float16* qkv  = (_Float16*)(smem + 16384);           // 6 x [64][64]
  _Float16* a0h  = qkv;                                 // alias (phase A/B)
  _Float16* wb0  = (_Float16*)(smem + 65536);           // weight buf 0 (16KB)
  _Float16* wb1  = (_Float16*)(smem + 81920);           // weight buf 1 (16KB)
  float*    sf   = (float*)(smem + 98304);              // [64][64] f32 scratch
  _Float16* ph   = (_Float16*)(smem + 114688);          // [64][64] probs
  float*    xf0f = (float*)(smem + 122880);             // [64][64] LN residual
  float*    red  = (float*)(smem + 139264);             // [64][2]

  const int tid  = threadIdx.x;
  const int wave = tid >> 5, lane = tid & 31;
  const int b    = blockIdx.x / nWF;
  const int wid  = sel[(size_t)b * nWF + (blockIdx.x % nWF)];
  const int wy   = (wid / nW) * 8 - padT;
  const int wx   = (wid % nW) * 8 - padL;
  const size_t cstride = (size_t)512 * 512;
  float* xbase = xg + (size_t)b * 64 * cstride;

  // prefetch lin_w -> wb0, qkv slice0 -> wb1 (4 async instr per wave each)
  stage_rows((const char*)lin16, 256, (char*)wb0, 256, 64, 256, tid);
  stage_rows((const char*)qkv16, 768, (char*)wb1, 128, 128, 128, tid);

  // load window, coalesced (consecutive tids -> consecutive x)
  for (int i = tid; i < 4096; i += 256) {
    int c = i >> 6, p = i & 63;
    int y = wy + (p >> 3), x = wx + (p & 7);
    float v = 0.f;
    if ((unsigned)y < 512u && (unsigned)x < 512u)
      v = xbase[(size_t)c * cstride + (size_t)y * 512 + x];
    sf[p * 64 + c] = v;
  }
  __syncthreads();

  if (tid < 64) {                                       // LN stats per pixel row
    float s = 0.f, s2 = 0.f;
    for (int c = 0; c < 64; ++c) { float v = sf[tid * 64 + c]; s += v; s2 += v * v; }
    float mu = s * (1.f / 64.f);
    red[tid * 2]     = mu;
    red[tid * 2 + 1] = rsqrtf(s2 * (1.f / 64.f) - mu * mu + 1e-5f);
  }
  __syncthreads();
  for (int i = tid; i < 4096; i += 256) {               // apply LN
    int p = i >> 6, c = i & 63;
    float v = (sf[p * 64 + c] - red[p * 2]) * red[p * 2 + 1] * lng[c] + lnb[c];
    xf0f[p * 64 + c] = v;
    a0h[p * 64 + c]  = (_Float16)v;
  }
  STAGE_WAIT(4);                                        // lin_w resident (slice0 in flight)
  __syncthreads();

  // GEMM1: xf = gelu(xf0 @ lin_w + lin_b), M64 K64 N128
  for (int tt = wave; tt < 32; tt += 8) {
    int mt = tt >> 3, nt = tt & 7;
    v8f acc = {};
#pragma unroll
    for (int ks = 0; ks < 2; ++ks) {
      v16h af = frag_a(a0h, 64, mt * 16, ks * 32, lane);
      v16h bf = frag_b(wb0, 128, ks * 32, nt * 16, lane);
      acc = WMMA_F16(af, bf, acc);
    }
    int n = nt * 16 + (lane & 15);
    float bias = linb[n];
    int mbase = mt * 16 + ((lane & 16) >> 1);
#pragma unroll
    for (int r = 0; r < 8; ++r)
      xfh[(mbase + r) * 128 + n] = (_Float16)gelu_f(acc[r] + bias);
  }
  __syncthreads();

  // qkv: 6 GEMMs M64 K128 N64, slice hg staged in buf[(hg+1)&1], double-buffered
  for (int hg = 0; hg < 6; ++hg) {
    _Float16* cur = (hg & 1) ? wb0 : wb1;
    if (hg < 5) {
      int h2 = (hg + 1) / 3, m2 = (hg + 1) % 3;
      stage_rows((const char*)(qkv16 + m2 * 128 + h2 * 64), 768,
                 (char*)((hg & 1) ? wb1 : wb0), 128, 128, 128, tid);
      STAGE_WAIT(4);                                    // slice hg resident
    } else {
      STAGE_WAIT(0);
    }
    __syncthreads();
    _Float16* dst = qkv + hg * 4096;
    for (int tt = wave; tt < 16; tt += 8) {
      int mt = tt >> 2, nt = tt & 3;
      v8f acc = {};
#pragma unroll
      for (int ks = 0; ks < 4; ++ks) {
        v16h af = frag_a(xfh, 128, mt * 16, ks * 32, lane);
        v16h bf = frag_b(cur, 64, ks * 32, nt * 16, lane);
        acc = WMMA_F16(af, bf, acc);
      }
      int n = nt * 16 + (lane & 15);
      int mbase = mt * 16 + ((lane & 16) >> 1);
#pragma unroll
      for (int r = 0; r < 8; ++r) dst[(mbase + r) * 64 + n] = (_Float16)acc[r];
    }
    __syncthreads();
  }

  // attention per head: S = scale*q k^T ; softmax fp32 ; O = P v ; xf += O
  const float ATT_SCALE = 0.17677669529663687f;         // (C/HEADS)^-0.5
  for (int h = 0; h < 2; ++h) {
    const _Float16* qh = qkv + (h * 3 + 0) * 4096;
    const _Float16* kh = qkv + (h * 3 + 1) * 4096;
    const _Float16* vh = qkv + (h * 3 + 2) * 4096;
    for (int tt = wave; tt < 16; tt += 8) {
      int mt = tt >> 2, nt = tt & 3;
      v8f acc = {};
#pragma unroll
      for (int ks = 0; ks < 2; ++ks) {
        v16h af = frag_a(qh, 64, mt * 16, ks * 32, lane);
        v16h bf = frag_bt(kh, 64, ks * 32, nt * 16, lane);
        acc = WMMA_F16(af, bf, acc);
      }
      int n = nt * 16 + (lane & 15);
      int mbase = mt * 16 + ((lane & 16) >> 1);
#pragma unroll
      for (int r = 0; r < 8; ++r) sf[(mbase + r) * 64 + n] = acc[r] * ATT_SCALE;
    }
    __syncthreads();
    if (tid < 64) {
      float mx = -3.4e38f;
      for (int j = 0; j < 64; ++j) mx = fmaxf(mx, sf[tid * 64 + j]);
      float sum = 0.f;
      for (int j = 0; j < 64; ++j) sum += __expf(sf[tid * 64 + j] - mx);
      float inv = 1.f / sum;
      for (int j = 0; j < 64; ++j)
        ph[tid * 64 + j] = (_Float16)(__expf(sf[tid * 64 + j] - mx) * inv);
    }
    __syncthreads();
    for (int tt = wave; tt < 16; tt += 8) {
      int mt = tt >> 2, nt = tt & 3;
      v8f acc = {};
#pragma unroll
      for (int ks = 0; ks < 2; ++ks) {
        v16h af = frag_a(ph, 64, mt * 16, ks * 32, lane);
        v16h bf = frag_b(vh, 64, ks * 32, nt * 16, lane);
        acc = WMMA_F16(af, bf, acc);
      }
      int n = nt * 16 + (lane & 15);
      int mbase = mt * 16 + ((lane & 16) >> 1);
#pragma unroll
      for (int r = 0; r < 8; ++r) sf[(mbase + r) * 64 + n] = acc[r];
    }
    __syncthreads();
    for (int i = tid; i < 4096; i += 256) {
      int p = i >> 6, d = i & 63;
      int e = p * 128 + h * 64 + d;
      xfh[e] = (_Float16)((float)xfh[e] + sf[i]);
    }
    __syncthreads();
  }

  // proj: out = gelu(xf @ proj_w + proj_b), M64 K128 N64
  stage_rows((const char*)proj16, 128, (char*)wb1, 128, 128, 128, tid);
  STAGE_WAIT(0);
  __syncthreads();
  for (int tt = wave; tt < 16; tt += 8) {
    int mt = tt >> 2, nt = tt & 3;
    v8f acc = {};
#pragma unroll
    for (int ks = 0; ks < 4; ++ks) {
      v16h af = frag_a(xfh, 128, mt * 16, ks * 32, lane);
      v16h bf = frag_b(wb1, 64, ks * 32, nt * 16, lane);
      acc = WMMA_F16(af, bf, acc);
    }
    int n = nt * 16 + (lane & 15);
    float bias = projb[n];
    int mbase = mt * 16 + ((lane & 16) >> 1);
#pragma unroll
    for (int r = 0; r < 8; ++r) sf[(mbase + r) * 64 + n] = gelu_f(acc[r] + bias);
  }
  __syncthreads();

  // scatter-add, coalesced; windows disjoint -> plain RMW
  for (int i = tid; i < 4096; i += 256) {
    int c = i >> 6, p = i & 63;
    int y = wy + (p >> 3), x = wx + (p & 7);
    if ((unsigned)y < 512u && (unsigned)x < 512u) {
      float* dptr = xbase + (size_t)c * cstride + (size_t)y * 512 + x;
      *dptr += xf0f[p * 64 + c] + sf[p * 64 + c];
    }
  }
}

// ---------------------------------------------------------------------------
// Host orchestration. d_in (depth-first insertion order):
//  0 feature_map, 1 uncertain_map, 2..5 conv_in{dw,pw,bn_g,bn_b},
//  6..17 convs[0..2]x4, 18..45 blocks[0..3]x{ln_g,ln_b,lin_w,lin_b,qkv_w,proj_w,proj_b}
// ws: [x ping buffer 128MB][scores][sel][f16 weights]
// ---------------------------------------------------------------------------
extern "C" void kernel_launch(void* const* d_in, const int* in_sizes, int n_in,
                              void* d_out, int out_size, void* d_ws, size_t ws_size,
                              hipStream_t stream) {
  (void)in_sizes; (void)n_in; (void)out_size; (void)ws_size;
  const float* feat = (const float*)d_in[0];
  const float* unc  = (const float*)d_in[1];
  float* bufA = (float*)d_ws;
  float* outp = (float*)d_out;
  const size_t xbytes = (size_t)2 * 64 * 512 * 512 * 4;
  float* scores = (float*)((char*)d_ws + xbytes);
  int*   sel    = (int*)(scores + 2 * 4225);
  _Float16* w16 = (_Float16*)((char*)d_ws + xbytes + 131072);
  // w16 layout (halfs): pw16[c]=c*4096 (4x); blocks at 16384+bk*65536:
  //   lin16 +0 (8192), qkv16 +8192 (49152), proj16 +57344 (8192)

  for (int c = 0; c < 4; ++c)
    cvt_pw_kernel<<<dim3(16), dim3(256), 0, stream>>>((const float*)d_in[3 + 4 * c],
                                                      w16 + c * 4096);
  for (int bk = 0; bk < 4; ++bk) {
    _Float16* dst = w16 + 16384 + bk * 65536;
    cvt_f16_kernel<<<dim3(32),  dim3(256), 0, stream>>>((const float*)d_in[18 + 7 * bk + 2], dst, 8192);
    cvt_f16_kernel<<<dim3(192), dim3(256), 0, stream>>>((const float*)d_in[18 + 7 * bk + 4], dst + 8192, 49152);
    cvt_f16_kernel<<<dim3(32),  dim3(256), 0, stream>>>((const float*)d_in[18 + 7 * bk + 5], dst + 57344, 8192);
  }

  auto conv = [&](const float* in, float* out, int ci, int res) {
    int pb = (ci == 0) ? 2 : 6 + 4 * (ci - 1);
    sepconv_kernel<<<dim3(2 * 64 * 64), dim3(256), (size_t)95488, stream>>>(
        in, (const float*)d_in[pb], w16 + ci * 4096,
        (const float*)d_in[pb + 2], (const float*)d_in[pb + 3], out, res);
  };
  auto refine = [&](float* xp, int bk, int shift) {
    int pb = 18 + 7 * bk;
    int padT = (shift >= 2) ? 4 : 0;
    int padL = (shift & 1) ? 4 : 0;
    int nH = (512 + 2 * padT) / 8, nW = (512 + 2 * padL) / 8;
    int nWin = nH * nW;
    int nWF = (int)(nWin * 0.3);                        // matches Python int() truncation
    _Float16* wblk = w16 + 16384 + bk * 65536;
    score_kernel<<<dim3(2 * nWin), dim3(64), 0, stream>>>(unc, scores, nH, nW, padT, padL);
    select_kernel<<<dim3(2), dim3(256), (size_t)nWin * 4, stream>>>(scores, sel, nWin, nWF);
    refine_kernel<<<dim3(2 * nWF), dim3(256), (size_t)139776, stream>>>(
        xp, sel, nW, nWF, padT, padL,
        (const float*)d_in[pb + 0], (const float*)d_in[pb + 1],
        wblk, (const float*)d_in[pb + 3],
        wblk + 8192, wblk + 57344, (const float*)d_in[pb + 6]);
  };

  conv(feat, bufA, 0, 0);       // conv_in, no residual
  refine(bufA, 0, 0);
  conv(bufA, outp, 1, 1);       // x + conv(x)
  refine(outp, 1, 1);
  conv(outp, bufA, 2, 2);
  refine(bufA, 2, 2);
  conv(bufA, outp, 3, 3);
  refine(outp, 3, 3);           // final result in d_out
}
